// Mamba2Block_38577396253299
// MI455X (gfx1250) — compile-verified, matrix operand order checked by reading, not run on hardware
//
#include <hip/hip_runtime.h>
#include <hip/hip_bf16.h>

typedef __attribute__((ext_vector_type(16))) __bf16 v16bf;
typedef __attribute__((ext_vector_type(8)))  __bf16 v8bf;
typedef __attribute__((ext_vector_type(8)))  float  v8f;

#define D_MODEL   768
#define D_INNER   1536
#define D_STATE   128
#define HEADDIM   64
#define NHEADS    24
#define CONV_DIM  1792
#define D_IN_PROJ 3352
#define BATCH     2
#define SEQLEN    2048
#define BL        (BATCH*SEQLEN)   /* 4096 */
#define EPS       1e-5f
#define CHUNK     32

// f32 -> bf16 bulk convert, 8 elements/thread (b128 in, b128 out)
extern "C" __global__ __launch_bounds__(256)
void cvt_bf16_kernel(const float* __restrict__ in, __bf16* __restrict__ out, int n8) {
  const int g = blockIdx.x * 256 + threadIdx.x;   // group of 8 elements
  if (g >= n8) return;
  const float4* q = (const float4*)(in + (size_t)g * 8);
  float4 a = q[0], b = q[1];
  v8bf r;
  r[0]=(__bf16)a.x; r[1]=(__bf16)a.y; r[2]=(__bf16)a.z; r[3]=(__bf16)a.w;
  r[4]=(__bf16)b.x; r[5]=(__bf16)b.y; r[6]=(__bf16)b.z; r[7]=(__bf16)b.w;
  *(v8bf*)(out + (size_t)g * 8) = r;
}

// Load one 16-bit WMMA operand fragment from a bf16 row-major matrix.
// ISA 7.12.2 (16-bit 16x32 layout): lane half (lane>>4) picks the K sub-range;
// element e -> k = 8*half + (e&7) + 16*(e>>3); each 8-group is contiguous
// => two 16-byte global_load_b128 per fragment, no converts.
__device__ __forceinline__ v16bf load_frag16(const __bf16* __restrict__ p) {
  v8bf lo = *(const v8bf*)p;          // k offsets 0..7
  v8bf hi = *(const v8bf*)(p + 16);   // k offsets 16..23
  v16bf r;
#pragma unroll
  for (int e = 0; e < 8; ++e) { r[e] = lo[e]; r[8 + e] = hi[e]; }
  return r;
}

// C[M,N] = A[M,K] * B[N,K]^T  (A,B bf16 row-major; C f32 row-major)
// block: 256 threads = 8 waves; wave tile 16Mx64N; block tile 64Mx128N.
extern "C" __global__ __launch_bounds__(256)
void gemm_nt_bf16(const __bf16* __restrict__ A, const __bf16* __restrict__ B,
                  float* __restrict__ C, int M, int N, int K) {
  const int lane = threadIdx.x & 31;
  const int w    = threadIdx.x >> 5;
  const int half = lane >> 4;
  const int l16  = lane & 15;
  const int m0   = blockIdx.x * 64  + (w & 3) * 16;
  const int n0   = blockIdx.y * 128 + (w >> 2) * 64;

  v8f acc[4];
#pragma unroll
  for (int j = 0; j < 4; ++j)
#pragma unroll
    for (int r = 0; r < 8; ++r) acc[j][r] = 0.0f;

  const __bf16* Abase = A + (size_t)(m0 + l16) * K + 8 * half;
  const __bf16* Bbase[4];
#pragma unroll
  for (int j = 0; j < 4; ++j) {
    int n = n0 + 16 * j + l16;
    n = (n < N) ? n : (N - 1);        // clamp: in-bounds load, column unused
    Bbase[j] = B + (size_t)n * K + 8 * half;
  }

#pragma unroll 2
  for (int kb = 0; kb < K; kb += 32) {
    const v16bf af = load_frag16(Abase + kb);
#pragma unroll
    for (int j = 0; j < 4; ++j) {
      const v16bf bf = load_frag16(Bbase[j] + kb);
      acc[j] = __builtin_amdgcn_wmma_f32_16x16x32_bf16(
          false, af, false, bf, (short)0, acc[j], false, false);
    }
  }

  // C/D layout: VGPR r -> M = r + 8*half, N = lane&15 (ISA 7.12.2)
#pragma unroll
  for (int j = 0; j < 4; ++j) {
    const int n = n0 + 16 * j + l16;
    if (n < N) {
#pragma unroll
      for (int r = 0; r < 8; ++r) {
        const int m = m0 + r + 8 * half;
        C[(size_t)m * N + n] = acc[j][r];
      }
    }
  }
}

// Depthwise causal conv (width 4) + bias + SiLU over xBC columns of zxbcdt.
extern "C" __global__ __launch_bounds__(256)
void conv_silu_kernel(const float* __restrict__ zx, const float* __restrict__ cw,
                      const float* __restrict__ cb, float* __restrict__ out) {
  const int idx = blockIdx.x * 256 + threadIdx.x;   // over BL*CONV_DIM
  if (idx >= BL * CONV_DIM) return;
  const int c  = idx % CONV_DIM;
  const int bl = idx / CONV_DIM;
  const int l  = bl % SEQLEN;
  const float* col = zx + (size_t)bl * D_IN_PROJ + D_INNER + c;
  float acc = cb[c];
#pragma unroll
  for (int j = 0; j < 4; ++j) {
    const int lp = l - 3 + j;
    if (lp >= 0) acc += col[(ptrdiff_t)(j - 3) * D_IN_PROJ] * cw[c * 4 + j];
  }
  out[idx] = acc / (1.0f + __expf(-acc));           // silu
}

// dt_dis = softplus(dt + dt_bias); dA = exp(dt_dis * (-exp(A_log)))
extern "C" __global__ __launch_bounds__(256)
void dtprep_kernel(const float* __restrict__ zx, const float* __restrict__ dt_bias,
                   const float* __restrict__ A_log, float* __restrict__ dtdis,
                   float* __restrict__ dAv) {
  const int idx = blockIdx.x * 256 + threadIdx.x;   // over BL*NHEADS
  if (idx >= BL * NHEADS) return;
  const int h  = idx % NHEADS;
  const int bl = idx / NHEADS;
  const float s  = zx[(size_t)bl * D_IN_PROJ + D_INNER + CONV_DIM + h] + dt_bias[h];
  const float sp = (s > 20.0f) ? s : log1pf(__expf(s));
  const float Ah = -__expf(A_log[h]);
  dtdis[idx] = sp;
  dAv[idx]   = __expf(sp * Ah);
}

// Sequential SSM scan. One block per (batch, head). State 64x128 f32 fully
// in registers (32 floats/thread). Chunk operands (B|C fused, x) are DMA'd
// straight to LDS with GLOBAL_LOAD_ASYNC_TO_LDS_B128 (ASYNCcnt), so the
// staging never touches VGPRs. Threads sharing p are 4 consecutive lanes of
// one wave32 -> shfl reduction for the C-contraction.
extern "C" __global__ __launch_bounds__(256)
void scan_kernel(const float* __restrict__ xbc, const float* __restrict__ dtdis,
                 const float* __restrict__ dAv, const float* __restrict__ Dp,
                 float* __restrict__ y) {
  __shared__ float sBC[CHUNK][2 * D_STATE];   // [t][0:128)=B, [t][128:256)=C
  __shared__ float sX[CHUNK][HEADDIM];
  __shared__ float sdA[CHUNK];
  __shared__ float sdt[CHUNK];

  const int bh  = blockIdx.x;
  const int b   = bh / NHEADS;
  const int h   = bh % NHEADS;
  const int tid = threadIdx.x;
  const int p   = tid >> 2;       // 0..63
  const int nq  = tid & 3;        // which 32-wide slice of D_STATE
  const int n0  = nq * 32;

  // LDS aperture: generic addr low 32 bits are the LDS byte offset (ISA 10.2)
  const uint32_t ldsBC = (uint32_t)(uintptr_t)&sBC[0][0];
  const uint32_t ldsX  = (uint32_t)(uintptr_t)&sX[0][0];

  float state[32];
#pragma unroll
  for (int i = 0; i < 32; ++i) state[i] = 0.0f;
  const float Dh = Dp[h];

  for (int c = 0; c < SEQLEN / CHUNK; ++c) {
    __syncthreads();
    const float* gBC = xbc + (size_t)(b * SEQLEN + c * CHUNK) * CONV_DIM + D_INNER;
    const float* gX  = xbc + (size_t)(b * SEQLEN + c * CHUNK) * CONV_DIM + h * HEADDIM;

    // B|C block: 32 rows x 1024B -> 2048 aligned 16B DMAs (8 per thread)
#pragma unroll
    for (int i = 0; i < 8; ++i) {
      const int idx = i * 256 + tid;
      const int tt  = idx >> 6;              // row (64 x 16B per row)
      const int c16 = idx & 63;
      const uint32_t lo = ldsBC + (uint32_t)idx * 16u;
      const uint32_t go = (uint32_t)(tt * CONV_DIM + c16 * 4) * 4u;
      asm volatile("global_load_async_to_lds_b128 %0, %1, %2"
                   :: "v"(lo), "v"(go), "s"(gBC) : "memory");
    }
    // x block: 32 rows x 256B -> 512 aligned 16B DMAs (2 per thread)
#pragma unroll
    for (int i = 0; i < 2; ++i) {
      const int idx = i * 256 + tid;
      const int tt  = idx >> 4;              // row (16 x 16B per row)
      const int c16 = idx & 15;
      const uint32_t lo = ldsX + (uint32_t)idx * 16u;
      const uint32_t go = (uint32_t)(tt * CONV_DIM + c16 * 4) * 4u;
      asm volatile("global_load_async_to_lds_b128 %0, %1, %2"
                   :: "v"(lo), "v"(go), "s"(gX) : "memory");
    }
    if (tid < CHUNK) {
      const int t = c * CHUNK + tid;
      sdA[tid] = dAv[(size_t)(b * SEQLEN + t) * NHEADS + h];
      sdt[tid] = dtdis[(size_t)(b * SEQLEN + t) * NHEADS + h];
    }
    asm volatile("s_wait_asynccnt 0x0" ::: "memory");
    __syncthreads();

    for (int tt = 0; tt < CHUNK; ++tt) {
      const float da   = sdA[tt];
      const float dtv  = sdt[tt];
      const float xv   = sX[tt][p];
      const float coef = dtv * xv;
      float acc = 0.0f;
#pragma unroll
      for (int i = 0; i < 32; ++i) {
        const float s = state[i] * da + coef * sBC[tt][n0 + i];
        state[i] = s;
        acc += s * sBC[tt][D_STATE + n0 + i];
      }
      acc += __shfl_xor(acc, 1, 32);
      acc += __shfl_xor(acc, 2, 32);
      if (nq == 0) {
        const int t = c * CHUNK + tt;
        y[(size_t)(b * SEQLEN + t) * D_INNER + h * HEADDIM + p] = acc + Dh * xv;
      }
    }
  }
}

// out = (y*silu(z)) * rsqrt(mean((y*silu(z))^2)+eps) * norm_w ; one block/row
extern "C" __global__ __launch_bounds__(256)
void gated_rmsnorm_kernel(const float* __restrict__ y, const float* __restrict__ zx,
                          const float* __restrict__ nw, float* __restrict__ out) {
  __shared__ float red[8];
  const int row = blockIdx.x;
  const int tid = threadIdx.x;
  const float* yr = y  + (size_t)row * D_INNER;
  const float* zr = zx + (size_t)row * D_IN_PROJ;
  float v[6];
  float ssum = 0.0f;
#pragma unroll
  for (int i = 0; i < 6; ++i) {
    const int d = tid + i * 256;
    const float zz = zr[d];
    const float yz = yr[d] * (zz / (1.0f + __expf(-zz)));
    v[i] = yz;
    ssum += yz * yz;
  }
#pragma unroll
  for (int o = 16; o > 0; o >>= 1) ssum += __shfl_xor(ssum, o, 32);
  if ((tid & 31) == 0) red[tid >> 5] = ssum;
  __syncthreads();
  if (tid == 0) {
    float t = 0.0f;
#pragma unroll
    for (int i = 0; i < 8; ++i) t += red[i];
    red[0] = rsqrtf(t / (float)D_INNER + EPS);
  }
  __syncthreads();
  const float r = red[0];
  float* orow = out + (size_t)row * D_INNER;
#pragma unroll
  for (int i = 0; i < 6; ++i) {
    const int d = tid + i * 256;
    orow[d] = v[i] * r * nw[d];
  }
}

extern "C" void kernel_launch(void* const* d_in, const int* in_sizes, int n_in,
                              void* d_out, int out_size, void* d_ws, size_t ws_size,
                              hipStream_t stream) {
  const float* u          = (const float*)d_in[0];
  const float* in_proj_w  = (const float*)d_in[1];
  const float* conv_w     = (const float*)d_in[2];
  const float* conv_b     = (const float*)d_in[3];
  const float* dt_bias    = (const float*)d_in[4];
  const float* A_log      = (const float*)d_in[5];
  const float* Dv         = (const float*)d_in[6];
  const float* norm_w     = (const float*)d_in[7];
  const float* out_proj_w = (const float*)d_in[8];
  float* out = (float*)d_out;

  // f32 scratch
  float* zx    = (float*)d_ws;                       // BL*D_IN_PROJ
  float* xbc   = zx    + (size_t)BL * D_IN_PROJ;     // BL*CONV_DIM
  float* dtdis = xbc   + (size_t)BL * CONV_DIM;      // BL*NHEADS
  float* dAv   = dtdis + (size_t)BL * NHEADS;        // BL*NHEADS
  float* ybuf  = dAv   + (size_t)BL * NHEADS;        // BL*D_INNER
  float* nbuf  = xbc;  // reuse conv buffer; conv data dead after scan
  // bf16 mirrors
  __bf16* ubf  = (__bf16*)(ybuf + (size_t)BL * D_INNER);       // BL*D_MODEL
  __bf16* w1bf = ubf  + (size_t)BL * D_MODEL;                  // D_IN_PROJ*D_MODEL
  __bf16* nbf  = w1bf + (size_t)D_IN_PROJ * D_MODEL;           // BL*D_INNER
  __bf16* w2bf = nbf  + (size_t)BL * D_INNER;                  // D_MODEL*D_INNER

  dim3 blk(256);

  cvt_bf16_kernel<<<(BL * D_MODEL / 8 + 255) / 256, blk, 0, stream>>>(u, ubf, BL * D_MODEL / 8);
  cvt_bf16_kernel<<<(D_IN_PROJ * D_MODEL / 8 + 255) / 256, blk, 0, stream>>>(in_proj_w, w1bf, D_IN_PROJ * D_MODEL / 8);
  cvt_bf16_kernel<<<(D_MODEL * D_INNER / 8 + 255) / 256, blk, 0, stream>>>(out_proj_w, w2bf, D_MODEL * D_INNER / 8);

  dim3 g1(BL / 64, (D_IN_PROJ + 127) / 128);
  gemm_nt_bf16<<<g1, blk, 0, stream>>>(ubf, w1bf, zx, BL, D_IN_PROJ, D_MODEL);

  conv_silu_kernel<<<(BL * CONV_DIM + 255) / 256, blk, 0, stream>>>(zx, conv_w, conv_b, xbc);

  dtprep_kernel<<<(BL * NHEADS + 255) / 256, blk, 0, stream>>>(zx, dt_bias, A_log, dtdis, dAv);

  scan_kernel<<<BATCH * NHEADS, blk, 0, stream>>>(xbc, dtdis, dAv, Dv, ybuf);

  gated_rmsnorm_kernel<<<BL, blk, 0, stream>>>(ybuf, zx, norm_w, nbuf);

  cvt_bf16_kernel<<<(BL * D_INNER / 8 + 255) / 256, blk, 0, stream>>>(nbuf, nbf, BL * D_INNER / 8);

  dim3 g2(BL / 64, D_MODEL / 128);
  gemm_nt_bf16<<<g2, blk, 0, stream>>>(nbf, w2bf, out, BL, D_MODEL, D_INNER);
}
